// LSTM_21268678050322
// MI455X (gfx1250) — matrix operand order repaired and checked by reading, not compiled, not run on hardware
//
#include <hip/hip_runtime.h>
#include <hip/hip_bf16.h>
#include <math.h>

// ---------------------------------------------------------------------------
// MI455X (gfx1250) 2-layer LSTM.
//  B=32, T=2048, D=128, H=512, 4H=2048.
//  Kernel 1: xg = X @ W.T + bih + bhh   (WMMA bf16 -> f32, LDS-staged tiles)
//  Kernel 2: persistent recurrence, Whh resident in LDS as bf16, per-step
//            M=32 x N=128 x K=512 WMMA GEMM per workgroup, grid barrier/step,
//            double-buffered h broadcast, async-to-LDS reload when available.
//  Kernel 3: final FC reduction.
// ---------------------------------------------------------------------------

typedef __attribute__((ext_vector_type(16))) __bf16 v16bf;
typedef __attribute__((ext_vector_type(8)))  float  v8f;
typedef __attribute__((ext_vector_type(4)))  int    v4i;

union V16BF { uint4 u[2]; v16bf v; };

#define LSTM_B   32
#define LSTM_T   2048
#define LSTM_H   512
#define LSTM_G4  2048            // 4*H
#define NWG_REC  16              // persistent workgroups in recurrent kernel

// LDS row pads (in bf16 elements) so 16 lanes hit distinct banks
#define GEMM_KPAD  (32 + 8)      // GEMM tiles: rows of 32 K-elems, padded
#define REC_KPAD   (512 + 8)     // recurrent: rows of 512 K-elems, padded

#define GEMM_SMEM  (2u * 128u * GEMM_KPAD * 2u)                        // 20480 B
#define REC_SMEM   ((128u*REC_KPAD + 32u*REC_KPAD) * 2u + 32u*128u*4u) // 182784 B

// Optional CDNA5 async global->LDS copy path (guarded: falls back to sync copy)
#if defined(__has_builtin)
#if __has_builtin(__builtin_amdgcn_global_load_async_to_lds_b128) && \
    __has_builtin(__builtin_amdgcn_s_wait_asynccnt)
#define HAVE_ASYNC_LDS 1
#endif
#endif

#ifdef HAVE_ASYNC_LDS
typedef __attribute__((address_space(1))) v4i* as1_v4i_ptr;   // global
typedef __attribute__((address_space(3))) v4i* as3_v4i_ptr;   // LDS
#endif

static __device__ __forceinline__ v8f wmma_bf16(const V16BF& a, const V16BF& b, v8f c) {
    return __builtin_amdgcn_wmma_f32_16x16x32_bf16(false, a.v, false, b.v,
                                                   (short)0, c, false, false);
}

// ---------------------------------------------------------------------------
// Kernel 1: out[M,N] = X[M,K] @ W[N,K]^T + b0[N] + b1[N]   (f32 in/out,
// bf16 WMMA with f32 accumulation). M%128==0, N%128==0, K%32==0.
// Block = 256 threads (8 waves), tile 128x128, wave tile 32x64.
// ---------------------------------------------------------------------------
__global__ void __launch_bounds__(256)
gemm_bias_kernel(const float* __restrict__ X, const float* __restrict__ W,
                 const float* __restrict__ b0, const float* __restrict__ b1,
                 float* __restrict__ out, int M, int N, int K)
{
    extern __shared__ char smem[];
    __bf16* sA = (__bf16*)smem;                               // [128][GEMM_KPAD]
    __bf16* sB = (__bf16*)(smem + 128 * GEMM_KPAD * 2);       // [128][GEMM_KPAD]

    const int tid  = threadIdx.x;
    const int nblk = blockIdx.x, mblk = blockIdx.y;
    const int w    = tid >> 5;
    const int lane = tid & 31;
    const int lm   = lane & 15;        // row-within-tile / col-within-tile
    const int lh   = lane >> 4;        // lane half selects K sub-range
    const int wm   = w & 3;            // wave M strip (4 strips of 32)
    const int wn   = w >> 2;           // wave N strip (2 strips of 64)

    v8f acc[2][4];
    const v8f vz = {0.f,0.f,0.f,0.f,0.f,0.f,0.f,0.f};
#pragma unroll
    for (int i = 0; i < 2; ++i)
#pragma unroll
        for (int j = 0; j < 4; ++j) acc[i][j] = vz;

    for (int kc = 0; kc < K; kc += 32) {
        // ---- stage A (128x32) and B (128x32), converting f32 -> bf16 ----
#pragma unroll
        for (int i = 0; i < 16; ++i) {
            int e = i * 256 + tid;            // 4096 elements
            int r = e >> 5, c = e & 31;
            sA[r * GEMM_KPAD + c] = (__bf16)X[(size_t)(mblk * 128 + r) * K + kc + c];
        }
#pragma unroll
        for (int i = 0; i < 16; ++i) {
            int e = i * 256 + tid;
            int r = e >> 5, c = e & 31;
            sB[r * GEMM_KPAD + c] = (__bf16)W[(size_t)(nblk * 128 + r) * K + kc + c];
        }
        __syncthreads();

        // ---- operand fetch: each lane = two aligned 16B LDS loads ----
        V16BF a[2], bb[4];
#pragma unroll
        for (int mt = 0; mt < 2; ++mt) {
            const uint4* q = (const uint4*)(sA + (wm * 32 + mt * 16 + lm) * GEMM_KPAD);
            a[mt].u[0] = q[lh];        // K 0..7   (lanes 0-15) / 8..15  (16-31)
            a[mt].u[1] = q[2 + lh];    // K 16..23 (lanes 0-15) / 24..31 (16-31)
        }
#pragma unroll
        for (int nt = 0; nt < 4; ++nt) {
            const uint4* q = (const uint4*)(sB + (wn * 64 + nt * 16 + lm) * GEMM_KPAD);
            bb[nt].u[0] = q[lh];           // K 0..7   / 8..15
            bb[nt].u[1] = q[2 + lh];       // K 16..23 / 24..31
        }
#pragma unroll
        for (int mt = 0; mt < 2; ++mt)
#pragma unroll
            for (int nt = 0; nt < 4; ++nt)
                acc[mt][nt] = wmma_bf16(a[mt], bb[nt], acc[mt][nt]);
        __syncthreads();
    }

    // ---- epilogue: C layout VGPR r -> (M = r + 8*lh, N = lm) ----
#pragma unroll
    for (int nt = 0; nt < 4; ++nt) {
        int col = nblk * 128 + wn * 64 + nt * 16 + lm;
        float bias = b0[col] + b1[col];
#pragma unroll
        for (int mt = 0; mt < 2; ++mt) {
            int row0 = mblk * 128 + wm * 32 + mt * 16 + lh * 8;
#pragma unroll
            for (int r = 0; r < 8; ++r)
                out[(size_t)(row0 + r) * N + col] = acc[mt][nt][r] + bias;
        }
    }
}

// ---------------------------------------------------------------------------
// Grid-wide barrier (NWG_REC co-resident workgroups), agent-scope atomics.
// ---------------------------------------------------------------------------
static __device__ __forceinline__ void grid_barrier(unsigned* cnt, unsigned* gen)
{
    __syncthreads();
    __threadfence();                      // release our hglob stores
    if (threadIdx.x == 0) {
        unsigned g = __hip_atomic_load(gen, __ATOMIC_RELAXED, __HIP_MEMORY_SCOPE_AGENT);
        unsigned arrived = __hip_atomic_fetch_add(cnt, 1u, __ATOMIC_ACQ_REL,
                                                  __HIP_MEMORY_SCOPE_AGENT);
        if (arrived == NWG_REC - 1) {
            __hip_atomic_store(cnt, 0u, __ATOMIC_RELAXED, __HIP_MEMORY_SCOPE_AGENT);
            __hip_atomic_fetch_add(gen, 1u, __ATOMIC_RELEASE, __HIP_MEMORY_SCOPE_AGENT);
        } else {
            while (__hip_atomic_load(gen, __ATOMIC_ACQUIRE, __HIP_MEMORY_SCOPE_AGENT) == g)
                __builtin_amdgcn_s_sleep(2);
        }
    }
    __syncthreads();
    __threadfence();                      // acquire: drop stale hglob lines
}

// ---------------------------------------------------------------------------
// Kernel 2: persistent LSTM recurrence for one layer.
//  16 WGs; WG owns h-cols [hc0, hc0+32) => gate rows {g*512 + hc0 + 0..31}.
//  Whh slice (128 x 512) lives in LDS as bf16 in WMMA-B order.
//  Per step: gates = h @ WhhT  (M=32,N=128,K=512) via 32 WMMA/wave, then
//  pointwise LSTM cell; c state in registers; h broadcast via L2 + barrier.
//  hglob is DOUBLE-BUFFERED by step parity to avoid cross-WG WAR races.
// ---------------------------------------------------------------------------
__global__ void __launch_bounds__(256)
lstm_recurrent_kernel(const float* __restrict__ xg, const float* __restrict__ Whh,
                      __bf16* __restrict__ hglob, float* __restrict__ hseq,
                      float* __restrict__ hlast, int store_seq, int T,
                      unsigned* bar_cnt, unsigned* bar_gen)
{
    extern __shared__ char smem[];
    __bf16* whh_s = (__bf16*)smem;                                   // [128][REC_KPAD]
    __bf16* h_a   = (__bf16*)(smem + 128 * REC_KPAD * 2);            // [32][REC_KPAD]
    float*  gbuf  = (float*)(smem + (128 + 32) * REC_KPAD * 2);      // [32][128]

    const int tid  = threadIdx.x;
    const int hc0  = blockIdx.x * 32;
    const int w    = tid >> 5;
    const int lane = tid & 31;
    const int lm   = lane & 15;
    const int lh   = lane >> 4;

    // ---- stage Whh slice into LDS (f32 -> bf16), rows = local gate index ----
    // local gate row nl in [0,128): global gate row gr = (nl>>5)*512 + hc0 + (nl&31)
#pragma unroll 4
    for (int i = 0; i < 256; ++i) {
        int e  = i * 256 + tid;            // 128*512 elements
        int nl = e >> 9, c = e & 511;
        int gr = (nl >> 5) * LSTM_H + hc0 + (nl & 31);
        whh_s[nl * REC_KPAD + c] = (__bf16)Whh[(size_t)gr * LSTM_H + c];
    }
    // ---- zero h operand buffer (h_0 = 0) ----
#pragma unroll 5
    for (int i = 0; i < 65; ++i) {
        int e = i * 256 + tid;             // 32*520 = 16640 elements
        h_a[e] = (__bf16)0.0f;
    }
    __syncthreads();

    // c state in registers: thread handles 4 (m, j) cells: e = p*256+tid
    float creg[4] = {0.f, 0.f, 0.f, 0.f};

    // gate column owned by this wave/lane: nl = w*16 + lm
    const int nl  = w * 16 + lm;
    const int gr  = (nl >> 5) * LSTM_H + hc0 + (nl & 31);   // global gate col

    for (int t = 0; t < T; ++t) {
        // ---- gates = h @ WhhT for our 16 columns, K = 512 ----
        v8f acc0 = {0.f,0.f,0.f,0.f,0.f,0.f,0.f,0.f};
        v8f acc1 = acc0;
        const uint4* qb_base = (const uint4*)(whh_s + nl * REC_KPAD);
#pragma unroll 4
        for (int kk = 0; kk < 16; ++kk) {
            V16BF a0, a1, bb;
            const uint4* qa0 = (const uint4*)(h_a + (0  + lm) * REC_KPAD + kk * 32);
            const uint4* qa1 = (const uint4*)(h_a + (16 + lm) * REC_KPAD + kk * 32);
            a0.u[0] = qa0[lh];     a0.u[1] = qa0[2 + lh];
            a1.u[0] = qa1[lh];     a1.u[1] = qa1[2 + lh];
            const uint4* qb = qb_base + kk * 4;
            bb.u[0] = qb[lh];      bb.u[1] = qb[2 + lh];
            acc0 = wmma_bf16(a0, bb, acc0);
            acc1 = wmma_bf16(a1, bb, acc1);
        }

        // ---- add precomputed input contribution, spill gates to LDS ----
        // C layout: VGPR r -> (M = mt*16 + lh*8 + r, N = lm)
        const float* xg_col = xg + (size_t)t * LSTM_G4 + gr;
#pragma unroll
        for (int r = 0; r < 8; ++r) {
            int m0 = lh * 8 + r;
            gbuf[m0 * 128 + nl] = acc0[r] + xg_col[(size_t)m0 * T * LSTM_G4];
            int m1 = 16 + lh * 8 + r;
            gbuf[m1 * 128 + nl] = acc1[r] + xg_col[(size_t)m1 * T * LSTM_G4];
        }
        if (t + 1 < T) {   // prefetch next timestep's gate inputs into L2
            __builtin_prefetch(xg_col + LSTM_G4, 0, 1);
        }
        __syncthreads();

        // ping-pong h broadcast buffer: step t publishes into buffer (t+1)&1
        __bf16* hg_w = hglob + (size_t)((t + 1) & 1) * (LSTM_B * LSTM_H);

        // ---- pointwise LSTM cell: thread owns 4 (m, j) cells ----
#pragma unroll
        for (int p = 0; p < 4; ++p) {
            int e = p * 256 + tid;
            int m = e >> 5, j = e & 31;
            float gi = gbuf[m * 128 +        j];
            float gf = gbuf[m * 128 + 32  +  j];
            float gg = gbuf[m * 128 + 64  +  j];
            float go = gbuf[m * 128 + 96  +  j];
            gi = 1.0f / (1.0f + __expf(-gi));
            gf = 1.0f / (1.0f + __expf(-gf));
            go = 1.0f / (1.0f + __expf(-go));
            gg = tanhf(gg);
            float c = gf * creg[p] + gi * gg;
            creg[p] = c;
            float h = go * tanhf(c);
            hg_w[m * LSTM_H + hc0 + j] = (__bf16)h;         // broadcast (bf16)
            if (store_seq)
                hseq[((size_t)m * T + t) * LSTM_H + hc0 + j] = h;
            else
                hlast[m * LSTM_H + hc0 + j] = h;
        }

        // ---- device-wide step barrier, then reload full h into LDS ----
        grid_barrier(bar_cnt, bar_gen);
        const uint4* hg4 = (const uint4*)hg_w;               // 2048 uint4
#ifdef HAVE_ASYNC_LDS
        // CDNA5 async copy: global -> LDS without touching VGPRs (ASYNCcnt)
#pragma unroll
        for (int i = 0; i < 8; ++i) {
            int e = i * 256 + tid;
            int m = e >> 6;                                  // 64 uint4 per row
            int k = (e & 63) * 8;
            __builtin_amdgcn_global_load_async_to_lds_b128(
                (as1_v4i_ptr)(hg4 + e),
                (as3_v4i_ptr)(h_a + m * REC_KPAD + k),
                0, 0);
        }
        __builtin_amdgcn_s_wait_asynccnt(0);
#else
#pragma unroll
        for (int i = 0; i < 8; ++i) {
            int e = i * 256 + tid;
            uint4 v = hg4[e];
            int m = e >> 6;                                  // 64 uint4 per row
            int k = (e & 63) * 8;
            *(uint4*)(h_a + m * REC_KPAD + k) = v;
        }
#endif
        __syncthreads();
    }
}

// ---------------------------------------------------------------------------
// Kernel 3: out[b] = dot(h2_last[b,:], Wfc[0,:]) + bfc[0]
// ---------------------------------------------------------------------------
__global__ void __launch_bounds__(64)
fc_kernel(const float* __restrict__ h, const float* __restrict__ Wfc,
          const float* __restrict__ bfc, float* __restrict__ out)
{
    __shared__ float red[64];
    int b = blockIdx.x, t = threadIdx.x;
    float s = 0.f;
    for (int k = t; k < LSTM_H; k += 64) s += h[b * LSTM_H + k] * Wfc[k];
    red[t] = s;
    __syncthreads();
    for (int off = 32; off > 0; off >>= 1) {
        if (t < off) red[t] += red[t + off];
        __syncthreads();
    }
    if (t == 0) out[b] = red[0] + bfc[0];
}

// ---------------------------------------------------------------------------
// Host orchestration
// ---------------------------------------------------------------------------
extern "C" void kernel_launch(void* const* d_in, const int* in_sizes, int n_in,
                              void* d_out, int out_size, void* d_ws, size_t ws_size,
                              hipStream_t stream)
{
    const float* x    = (const float*)d_in[0];
    const float* Wih0 = (const float*)d_in[1];
    const float* Whh0 = (const float*)d_in[2];
    const float* bih0 = (const float*)d_in[3];
    const float* bhh0 = (const float*)d_in[4];
    const float* Wih1 = (const float*)d_in[5];
    const float* Whh1 = (const float*)d_in[6];
    const float* bih1 = (const float*)d_in[7];
    const float* bhh1 = (const float*)d_in[8];
    const float* Wfc  = (const float*)d_in[9];
    const float* bfc  = (const float*)d_in[10];

    const size_t M  = (size_t)LSTM_B * LSTM_T;           // 65536
    const size_t XG = M * LSTM_G4;                       // xg elems
    const size_t H1 = M * LSTM_H;                        // h1 sequence elems

    char* ws = (char*)d_ws;
    size_t off = 0;
    float*          xg     = (float*)(ws + off);  off += XG * sizeof(float);
    float*          h1     = (float*)(ws + off);  off += H1 * sizeof(float);
    __bf16*         hglob  = (__bf16*)(ws + off); off += (size_t)2 * LSTM_B * LSTM_H * 2;
    float*          h2last = (float*)(ws + off);  off += (size_t)LSTM_B * LSTM_H * 4;
    unsigned*       bar    = (unsigned*)(ws + off); off += 256;

    (void)hipMemsetAsync(bar, 0, 2 * sizeof(unsigned), stream);
    (void)hipFuncSetAttribute((const void*)lstm_recurrent_kernel,
                              hipFuncAttributeMaxDynamicSharedMemorySize, REC_SMEM);

    dim3 ggrid(LSTM_G4 / 128, (unsigned)(M / 128));      // (16, 512)

    // layer 0: xg0 = x @ Wih0.T + bih0 + bhh0   (K = 128)
    gemm_bias_kernel<<<ggrid, 256, GEMM_SMEM, stream>>>(
        x, Wih0, bih0, bhh0, xg, (int)M, LSTM_G4, 128);
    lstm_recurrent_kernel<<<NWG_REC, 256, REC_SMEM, stream>>>(
        xg, Whh0, hglob, h1, h2last, 1, LSTM_T, bar, bar + 1);

    // layer 1: xg1 = h1 @ Wih1.T + bih1 + bhh1  (K = 512)
    gemm_bias_kernel<<<ggrid, 256, GEMM_SMEM, stream>>>(
        h1, Wih1, bih1, bhh1, xg, (int)M, LSTM_G4, 512);
    lstm_recurrent_kernel<<<NWG_REC, 256, REC_SMEM, stream>>>(
        xg, Whh1, hglob, h1, h2last, 0, LSTM_T, bar, bar + 1);

    fc_kernel<<<LSTM_B, 64, 0, stream>>>(h2last, Wfc, bfc, (float*)d_out);
}
